// ConvBaseline_88450556494469
// MI455X (gfx1250) — compile-verified, matrix operand order checked by reading, not run on hardware
//
#include <hip/hip_runtime.h>
#include <hip/hip_bf16.h>
#include <math.h>

// ---------------------------------------------------------------------------
// Fused neural-operator step for MI455X (gfx1250, wave32, WMMA).
// 1 prep kernel (weight bf16 swizzle), 1 encoder kernel, 32 fused step
// kernels. Each step-kernel block owns a 128-token x-tile (halo 6), keeps psi
// in LDS across 3 depths + decoder; all matmuls via v_wmma_f32_16x16x32_bf16.
// B-fragments are batch-loaded ahead of each WMMA burst so ds_load latency
// overlaps matrix issue (avoids per-WMMA s_wait_dscnt 0 drains).
// ---------------------------------------------------------------------------

#define H 64
#define H2 128
#define XT 128
#define EXT 144      // 128 + 2*HALO(6) rounded to 9 wmma row tiles
#define NTILE 9
#define XDIM 8192
#define BDIM 32
#define TOUT 32

typedef __attribute__((ext_vector_type(16))) __bf16 v16bf;
typedef __attribute__((ext_vector_type(8)))  float  v8f;

union ABfrag { v16bf v; uint4 q[2]; };

__device__ __forceinline__ unsigned short f2bf(float f) {
  unsigned int u = __float_as_uint(f);
  unsigned int r = (u + 0x7FFFu + ((u >> 16) & 1u)) >> 16;  // RNE
  return (unsigned short)r;
}
__device__ __forceinline__ float gelu_exact(float v) {
  return 0.5f * v * (1.0f + erff(v * 0.70710678118654752f));
}

// LDS byte offsets (dynamic LDS, total 186880 B)
#define OFF_PSI 0        // float[144*64]            36864 B
#define OFF_ACT 36864    // ushort[148*64] (conv in) 18944 B
#define OFF_H   55808    // ushort[144*64]           18432 B
#define OFF_M   74240    // ushort[144*128] / float[144*64] alias  36864 B
#define OFF_W   111104   // swizzled weights         73728 B
#define OFF_C   184832   // float[512] consts         2048 B
#define SMEM_BYTES 186880
// wbuf element (ushort) offsets
#define W1OFF 20480
#define W2OFF 28672

__device__ __forceinline__ void loadB(ABfrag& f, const unsigned short* bp) {
  f.q[0] = *(const uint4*)bp;
  f.q[1] = *(const uint4*)(bp + 8);
}
__device__ __forceinline__ void loadA(ABfrag& f, const unsigned short* ap) {
  f.q[0] = *(const uint4*)ap;
  f.q[1] = *(const uint4*)(ap + 16);
}

// ---------------------------------------------------------------------------
// Weight pre-swizzle: fp32 -> bf16, laid out as WMMA B-fragments (32x16):
// lane = hi(k)*16 + (n&15), 16 contiguous bf16 per lane: elems 0..7 = K chunk
// [hi*8, hi*8+8), elems 8..15 = K chunk [hi*8+16, hi*8+24)  (mirrors doc'd
// 16-bit A layout). So in-kernel B loads are two ds_load_b128 per lane.
// ---------------------------------------------------------------------------
__global__ void NOStep_prep_kernel(const float* __restrict__ conv_w,
                                   const float* __restrict__ mlp_w1,
                                   const float* __restrict__ mlp_w2,
                                   const float* __restrict__ dec_w1,
                                   unsigned short* __restrict__ conv_sw,
                                   unsigned short* __restrict__ w1_sw,
                                   unsigned short* __restrict__ w2_sw,
                                   unsigned short* __restrict__ dec_sw) {
  int id = blockIdx.x * 256 + threadIdx.x;
  if (id < 3 * 20480) {                      // conv_w [3][o=64][i=64][k=5]
    int d = id / 20480, e = id % 20480;
    int k = e % 5, i2 = (e / 5) % 64, o = e / 320;
    int lane = (((i2 >> 3) & 1) << 4) | (o & 15);
    int elem = ((i2 >> 4) & 1) * 8 + (i2 & 7);
    int frag = (k * 2 + (i2 >> 5)) * 4 + (o >> 4);
    conv_sw[d * 20480 + frag * 512 + lane * 16 + elem] = f2bf(conv_w[id]);
    return;
  }
  id -= 3 * 20480;
  if (id < 3 * 8192) {                       // mlp_w1 [3][f=128][d=64] -> B[k=d][n=f]
    int d = id / 8192, e = id % 8192;
    int f = e / 64, dd = e % 64;
    int lane = (((dd >> 3) & 1) << 4) | (f & 15);
    int elem = ((dd >> 4) & 1) * 8 + (dd & 7);
    int frag = (dd >> 5) * 8 + (f >> 4);
    w1_sw[d * 8192 + frag * 512 + lane * 16 + elem] = f2bf(mlp_w1[d * 8192 + e]);
    return;
  }
  id -= 3 * 8192;
  if (id < 3 * 8192) {                       // mlp_w2 [3][o=64][f=128] -> B[k=f][n=o]
    int d = id / 8192, e = id % 8192;
    int o = e / 128, f = e % 128;
    int lane = (((f >> 3) & 1) << 4) | (o & 15);
    int elem = ((f >> 4) & 1) * 8 + (f & 7);
    int frag = (f >> 5) * 4 + (o >> 4);
    w2_sw[d * 8192 + frag * 512 + lane * 16 + elem] = f2bf(mlp_w2[d * 8192 + e]);
    return;
  }
  id -= 3 * 8192;
  if (id < 4096) {                           // dec_w1 [o=64][i=64] -> B[k=i][n=o]
    int o = id / 64, i2 = id % 64;
    int lane = (((i2 >> 3) & 1) << 4) | (o & 15);
    int elem = ((i2 >> 4) & 1) * 8 + (i2 & 7);
    int frag = (i2 >> 5) * 4 + (o >> 4);
    dec_sw[frag * 512 + lane * 16 + elem] = f2bf(dec_w1[id]);
  }
}

// ---------------------------------------------------------------------------
// Encoder: psi[b,x,h] = enc(x) then per-token (mean, ddof=1 std) normalize.
// ---------------------------------------------------------------------------
__global__ __launch_bounds__(256) void NOStep_enc_kernel(
    const float* __restrict__ x, const float* __restrict__ ew,
    const float* __restrict__ eb, float* __restrict__ psiOut) {
  int tok = blockIdx.x * 256 + threadIdx.x;     // 0 .. 262143
  int b = tok >> 13, t = tok & (XDIM - 1);
  float xv[16];
#pragma unroll
  for (int j = 0; j < 16; j++) xv[j] = x[((size_t)(b * 16 + j)) * XDIM + t];
  float a[H];
  float s = 0.f;
#pragma unroll
  for (int h = 0; h < H; h++) {
    float acc = eb[h];
#pragma unroll
    for (int j = 0; j < 16; j++) acc += xv[j] * ew[h * 16 + j];
    a[h] = acc; s += acc;
  }
  float mean = s * (1.0f / H);
  float q = 0.f;
#pragma unroll
  for (int h = 0; h < H; h++) { float d = a[h] - mean; q += d * d; }
  float inv = 1.0f / (sqrtf(q * (1.0f / 63.0f)) + 1e-6f);
#pragma unroll
  for (int h = 0; h < H; h++) psiOut[(size_t)tok * H + h] = (a[h] - mean) * inv;
}

// ---------------------------------------------------------------------------
// Fused step: 3x (conv5 -> MLP(gelu) -> residual+LN+clip) + decoder head.
// ---------------------------------------------------------------------------
__global__ __launch_bounds__(256, 1) void NOStep_step_kernel(
    const float* __restrict__ psiIn, float* __restrict__ psiOut,
    const unsigned short* __restrict__ conv_sw,
    const unsigned short* __restrict__ w1_sw,
    const unsigned short* __restrict__ w2_sw,
    const unsigned short* __restrict__ dec_sw,
    const float* __restrict__ conv_b, const float* __restrict__ mlp_b1,
    const float* __restrict__ mlp_b2, const float* __restrict__ ln_g,
    const float* __restrict__ ln_b, const float* __restrict__ dec_b1,
    const float* __restrict__ dec_w2, const float* __restrict__ dec_b2,
    float* __restrict__ out, int step) {
  extern __shared__ __align__(16) unsigned char smem[];
  float*          psi  = (float*)(smem + OFF_PSI);           // [144][64]
  unsigned short* act  = (unsigned short*)(smem + OFF_ACT);  // [148][64]
  unsigned short* hbuf = (unsigned short*)(smem + OFF_H);    // [144][64]
  unsigned short* mbuf = (unsigned short*)(smem + OFF_M);    // [144][128]
  unsigned short* wbuf = (unsigned short*)(smem + OFF_W);
  float*          cv   = (float*)(smem + OFF_C);

  const int tid = threadIdx.x, lane = tid & 31, wid = tid >> 5;
  const int nlo = lane & 15, hi = lane >> 4;
  const int b = blockIdx.y, x0 = blockIdx.x * XT;

  // ---- load psi tile + halo (rows 0..139 map to tokens x0-6..x0+133) ----
  for (int v = tid; v < EXT * 16; v += 256) {
    int r = v >> 4, c = v & 15;
    int t = x0 + r - 6;
    float4 val;
    if (r < 140 && t >= 0 && t < XDIM)
      val = *(const float4*)(psiIn + ((size_t)(b * XDIM + t)) * H + c * 4);
    else
      val = make_float4(0.f, 0.f, 0.f, 0.f);
    *(float4*)(psi + r * H + c * 4) = val;
  }
  __syncthreads();

  for (int d = 0; d < 3; d++) {
    // ---- act = bf16(psi) with +2 row offset so conv taps never go negative
    for (int idx = tid; idx < 148 * 32; idx += 256) {
      int a = idx >> 5, hp = (idx & 31) * 2;
      int r = a - 2;
      float v0 = 0.f, v1 = 0.f;
      if (r >= 0 && r < EXT) { v0 = psi[r * H + hp]; v1 = psi[r * H + hp + 1]; }
      ((unsigned int*)act)[idx] = (unsigned)f2bf(v0) | ((unsigned)f2bf(v1) << 16);
    }
    // ---- stage this depth's swizzled weights + biases into LDS
    {
      uint4* wb4 = (uint4*)wbuf;
      const uint4* cs = (const uint4*)(conv_sw + d * 20480);
      for (int i = tid; i < 2560; i += 256) wb4[i] = cs[i];
      const uint4* s1 = (const uint4*)(w1_sw + d * 8192);
      for (int i = tid; i < 1024; i += 256) wb4[2560 + i] = s1[i];
      const uint4* s2 = (const uint4*)(w2_sw + d * 8192);
      for (int i = tid; i < 1024; i += 256) wb4[3584 + i] = s2[i];
      for (int i = tid; i < 384; i += 256) {
        float v;
        if (i < 64)       v = conv_b[d * 64 + i];
        else if (i < 192) v = mlp_b1[d * 128 + (i - 64)];
        else if (i < 256) v = mlp_b2[d * 64 + (i - 192)];
        else if (i < 320) v = ln_g[d * 64 + (i - 256)];
        else              v = ln_b[d * 64 + (i - 320)];
        cv[i] = v;
      }
    }
    __syncthreads();

    // ================= CONV (5 shifted WMMA GEMMs, K=64) =================
    for (int mt = wid; mt < NTILE; mt += 8) {
      const int rowA = mt * 16 + nlo;
      v8f acc[4];
#pragma unroll
      for (int ng = 0; ng < 4; ng++) {
        float bv = cv[ng * 16 + nlo];
#pragma unroll
        for (int i = 0; i < 8; i++) acc[ng][i] = bv;
      }
#pragma unroll
      for (int tap = 0; tap < 5; tap++) {
#pragma unroll
        for (int kg = 0; kg < 2; kg++) {
          // batch-load A + 4 B fragments, THEN issue 4 WMMAs (overlap dscnt)
          ABfrag a, bf[4];
          loadA(a, act + (rowA + tap) * H + kg * 32 + hi * 8);
          int fb = (tap * 2 + kg) * 4;
#pragma unroll
          for (int ng = 0; ng < 4; ng++)
            loadB(bf[ng], wbuf + (fb + ng) * 512 + lane * 16);
#pragma unroll
          for (int ng = 0; ng < 4; ng++)
            acc[ng] = __builtin_amdgcn_wmma_f32_16x16x32_bf16(
                false, a.v, false, bf[ng].v, (short)0, acc[ng], false, false);
        }
      }
#pragma unroll
      for (int ng = 0; ng < 4; ng++)
#pragma unroll
        for (int i = 0; i < 8; i++)
          hbuf[(mt * 16 + i + hi * 8) * H + ng * 16 + nlo] = f2bf(acc[ng][i]);
    }
    __syncthreads();

    // ================= MLP1: [tok,64]x[64,128] + GELU =================
    for (int mt = wid; mt < NTILE; mt += 8) {
      const int rowA = mt * 16 + nlo;
      v8f acc[8];
#pragma unroll
      for (int ng = 0; ng < 8; ng++) {
        float bv = cv[64 + ng * 16 + nlo];
#pragma unroll
        for (int i = 0; i < 8; i++) acc[ng][i] = bv;
      }
#pragma unroll
      for (int kg = 0; kg < 2; kg++) {
        ABfrag a;
        loadA(a, hbuf + rowA * H + kg * 32 + hi * 8);
#pragma unroll
        for (int half = 0; half < 2; half++) {   // 2x(4 B frags + 4 WMMAs)
          ABfrag bf[4];
#pragma unroll
          for (int j = 0; j < 4; j++)
            loadB(bf[j], wbuf + W1OFF + (kg * 8 + half * 4 + j) * 512 + lane * 16);
#pragma unroll
          for (int j = 0; j < 4; j++)
            acc[half * 4 + j] = __builtin_amdgcn_wmma_f32_16x16x32_bf16(
                false, a.v, false, bf[j].v, (short)0, acc[half * 4 + j], false, false);
        }
      }
#pragma unroll
      for (int ng = 0; ng < 8; ng++)
#pragma unroll
        for (int i = 0; i < 8; i++)
          mbuf[(mt * 16 + i + hi * 8) * H2 + ng * 16 + nlo] =
              f2bf(gelu_exact(acc[ng][i]));
    }
    __syncthreads();

    // ================= MLP2: [tok,128]x[128,64] + residual =================
    for (int mt = wid; mt < NTILE; mt += 8) {
      const int rowA = mt * 16 + nlo;
      v8f acc[4];
#pragma unroll
      for (int ng = 0; ng < 4; ng++) {
        float bv = cv[192 + ng * 16 + nlo];
#pragma unroll
        for (int i = 0; i < 8; i++) acc[ng][i] = bv;
      }
#pragma unroll
      for (int kg = 0; kg < 4; kg++) {
        ABfrag a, bf[4];
        loadA(a, mbuf + rowA * H2 + kg * 32 + hi * 8);
#pragma unroll
        for (int ng = 0; ng < 4; ng++)
          loadB(bf[ng], wbuf + W2OFF + (kg * 4 + ng) * 512 + lane * 16);
#pragma unroll
        for (int ng = 0; ng < 4; ng++)
          acc[ng] = __builtin_amdgcn_wmma_f32_16x16x32_bf16(
              false, a.v, false, bf[ng].v, (short)0, acc[ng], false, false);
      }
#pragma unroll
      for (int ng = 0; ng < 4; ng++)
#pragma unroll
        for (int i = 0; i < 8; i++) {
          int idx = (mt * 16 + i + hi * 8) * H + ng * 16 + nlo;
          psi[idx] += acc[ng][i];
        }
    }
    __syncthreads();

    // ============== LayerNorm + clip (wave32 butterfly reduce) ==============
    for (int r = wid; r < EXT; r += 8) {
      float v0 = psi[r * H + lane], v1 = psi[r * H + lane + 32];
      float s = v0 + v1, q = v0 * v0 + v1 * v1;
#pragma unroll
      for (int m = 16; m >= 1; m >>= 1) {
        s += __shfl_xor(s, m, 32);
        q += __shfl_xor(q, m, 32);
      }
      float mean = s * (1.0f / H);
      float var = q * (1.0f / H) - mean * mean;
      float rstd = rsqrtf(var + 1e-5f);
      v0 = (v0 - mean) * rstd * cv[256 + lane] + cv[320 + lane];
      v1 = (v1 - mean) * rstd * cv[256 + lane + 32] + cv[320 + lane + 32];
      psi[r * H + lane]      = fminf(10.f, fmaxf(-10.f, v0));
      psi[r * H + lane + 32] = fminf(10.f, fmaxf(-10.f, v1));
    }
    __syncthreads();
  }

  // ======================== Decoder head ========================
  for (int idx = tid; idx < EXT * 32; idx += 256) {
    int r = idx >> 5, hp = (idx & 31) * 2;
    ((unsigned int*)hbuf)[idx] =
        (unsigned)f2bf(psi[r * H + hp]) | ((unsigned)f2bf(psi[r * H + hp + 1]) << 16);
  }
  {
    uint4* wb4 = (uint4*)wbuf;
    const uint4* sd = (const uint4*)dec_sw;
    for (int i = tid; i < 512; i += 256) wb4[i] = sd[i];
    if (tid < 64)       cv[tid] = dec_b1[tid];
    else if (tid < 128) cv[tid] = dec_w2[tid - 64];
    else if (tid == 128) cv[128] = dec_b2[0];
  }
  __syncthreads();

  float* yb = (float*)mbuf;  // alias: [144][64] fp32 gelu(dec1) output
  for (int mt = wid; mt < NTILE; mt += 8) {
    const int rowA = mt * 16 + nlo;
    v8f acc[4];
#pragma unroll
    for (int ng = 0; ng < 4; ng++) {
      float bv = cv[ng * 16 + nlo];
#pragma unroll
      for (int i = 0; i < 8; i++) acc[ng][i] = bv;
    }
#pragma unroll
    for (int kg = 0; kg < 2; kg++) {
      ABfrag a, bf[4];
      loadA(a, hbuf + rowA * H + kg * 32 + hi * 8);
#pragma unroll
      for (int ng = 0; ng < 4; ng++)
        loadB(bf[ng], wbuf + (kg * 4 + ng) * 512 + lane * 16);
#pragma unroll
      for (int ng = 0; ng < 4; ng++)
        acc[ng] = __builtin_amdgcn_wmma_f32_16x16x32_bf16(
            false, a.v, false, bf[ng].v, (short)0, acc[ng], false, false);
    }
#pragma unroll
    for (int ng = 0; ng < 4; ng++)
#pragma unroll
      for (int i = 0; i < 8; i++)
        yb[(mt * 16 + i + hi * 8) * H + ng * 16 + nlo] = gelu_exact(acc[ng][i]);
  }
  __syncthreads();

  // y = yb . dec_w2 + dec_b2 for interior tokens
  for (int r = wid; r < EXT; r += 8) {
    float p = yb[r * H + lane] * cv[64 + lane] + yb[r * H + lane + 32] * cv[96 + lane];
#pragma unroll
    for (int m = 16; m >= 1; m >>= 1) p += __shfl_xor(p, m, 32);
    if (lane == 0 && r >= 6 && r < 134)
      out[((size_t)(b * TOUT + step)) * XDIM + x0 + r - 6] = p + cv[128];
  }
  // store interior psi for next step
  for (int v = tid; v < XT * 16; v += 256) {
    int r = 6 + (v >> 4), c = v & 15;
    *(float4*)(psiOut + ((size_t)(b * XDIM + x0 + r - 6)) * H + c * 4) =
        *(float4*)(psi + r * H + c * 4);
  }
}

// ---------------------------------------------------------------------------
extern "C" void kernel_launch(void* const* d_in, const int* in_sizes, int n_in,
                              void* d_out, int out_size, void* d_ws, size_t ws_size,
                              hipStream_t stream) {
  (void)in_sizes; (void)n_in; (void)out_size; (void)ws_size;
  const float* x      = (const float*)d_in[0];
  const float* enc_w  = (const float*)d_in[1];
  const float* enc_b  = (const float*)d_in[2];
  const float* conv_w = (const float*)d_in[3];
  const float* conv_b = (const float*)d_in[4];
  const float* mlp_w1 = (const float*)d_in[5];
  const float* mlp_b1 = (const float*)d_in[6];
  const float* mlp_w2 = (const float*)d_in[7];
  const float* mlp_b2 = (const float*)d_in[8];
  const float* ln_g   = (const float*)d_in[9];
  const float* ln_b   = (const float*)d_in[10];
  const float* dec_w1 = (const float*)d_in[11];
  const float* dec_b1 = (const float*)d_in[12];
  const float* dec_w2 = (const float*)d_in[13];
  const float* dec_b2 = (const float*)d_in[14];
  float* out = (float*)d_out;

  char* ws = (char*)d_ws;
  float* psiA = (float*)ws;                                  // 64 MB
  float* psiB = (float*)(ws + 67108864ull);                  // 64 MB
  unsigned short* conv_sw = (unsigned short*)(ws + 134217728ull);
  unsigned short* w1_sw  = conv_sw + 3 * 20480;
  unsigned short* w2_sw  = w1_sw + 3 * 8192;
  unsigned short* dec_sw = w2_sw + 3 * 8192;

  NOStep_prep_kernel<<<448, 256, 0, stream>>>(conv_w, mlp_w1, mlp_w2, dec_w1,
                                              conv_sw, w1_sw, w2_sw, dec_sw);
  NOStep_enc_kernel<<<1024, 256, 0, stream>>>(x, enc_w, enc_b, psiA);

  float* pin = psiA;
  float* pout = psiB;
  for (int s = 0; s < TOUT; s++) {
    NOStep_step_kernel<<<dim3(XDIM / XT, BDIM), 256, SMEM_BYTES, stream>>>(
        pin, pout, conv_sw, w1_sw, w2_sw, dec_sw, conv_b, mlp_b1, mlp_b2,
        ln_g, ln_b, dec_b1, dec_w2, dec_b2, out, s);
    float* tmp = pin; pin = pout; pout = tmp;
  }
}
